// NestedFeedForward_36610301231252
// MI455X (gfx1250) — matrix operand order repaired and checked by reading, not compiled, not run on hardware
//
#include <hip/hip_runtime.h>

// ---------------------------------------------------------------------------
// NestedFeedForward for MI455X (gfx1250, wave32, WMMA, async global->LDS).
//
//   convert:  Xbf  = bf16( x masked to D_m input features )      (f32 -> bf16)
//             W1bf = bf16(w1), W2bf = bf16(w2)                    (L2-resident)
//   pass1:    H = gelu( Xbf @ W1bf^T + b1 )                       -> bf16 ws
//   pass2:    Y = ( H @ W2bf^T + b2 ) masked to D_m outputs       -> f32 out
//
// GEMM hot loop: software-pipelined (unroll-by-2, peeled prologue/tail)
// GLOBAL_LOAD_ASYNC_TO_LDS_B128 in GVS mode (SGPR base + 32-bit VGPR offset),
// static s_wait_asynccnt immediates, feeding v_wmma_f32_16x16x32_bf16 from
// ds_load_b128 fragments laid out per the ISA VGPR tables.
// ---------------------------------------------------------------------------

typedef __attribute__((ext_vector_type(16))) __bf16 v16bf;
typedef __attribute__((ext_vector_type(8)))  float  v8f;

#define D_MODEL     2048
#define NUM_EXPERTS 4
#define M_TOTAL     (4 * 4096)       // B * S tokens

#define BM 128
#define BN 128
#define BK 64
#define LDS_STRIDE (BK + 8)          // bf16 elems; 144 B rows keep 16B alignment
#define NTHREADS 256                 // 8 wave32s
#define NTILES   (D_MODEL / BK)      // 32 K-tiles

// f32 -> bf16, round-to-nearest-even
__device__ __forceinline__ unsigned short f2bf(float f) {
  unsigned int u = __builtin_bit_cast(unsigned int, f);
  u += 0x7FFFu + ((u >> 16) & 1u);
  return (unsigned short)(u >> 16);
}

__device__ __forceinline__ int dm_of(int m) {
  if (m < 1 || m > NUM_EXPERTS) return 0;
  return D_MODEL >> (NUM_EXPERTS - m);
}

// ---------------- CDNA5 async global->LDS + ASYNCcnt helpers ----------------

// GVS mode: mem_addr = SGPR_base + VGPR_offset32; LDS[VDST.lane] gets 16B.
__device__ __forceinline__ void async_ld_b128_gvs(unsigned int lds_addr,
                                                  unsigned int voff,
                                                  unsigned long long sbase) {
  asm volatile("global_load_async_to_lds_b128 %0, %1, %2"
               :: "v"(lds_addr), "v"(voff), "s"(sbase)
               : "memory");
}

__device__ __forceinline__ void wait_async_le8() {
#if __has_builtin(__builtin_amdgcn_s_wait_asynccnt)
  __builtin_amdgcn_s_wait_asynccnt(8);
#else
  asm volatile("s_wait_asynccnt 0x8" ::: "memory");
#endif
}

__device__ __forceinline__ void wait_async_le0() {
#if __has_builtin(__builtin_amdgcn_s_wait_asynccnt)
  __builtin_amdgcn_s_wait_asynccnt(0);
#else
  asm volatile("s_wait_asynccnt 0x0" ::: "memory");
#endif
}

// ---------------- WMMA fragment loads (ISA VGPR layouts) --------------------

struct Bits256 { unsigned int r[8]; };

// A fragment (16x32 bf16): lanes 0-15 hold K[kk+0..7] and K[kk+16..23];
// lanes 16-31 hold K[kk+8..15] and K[kk+24..31].
__device__ __forceinline__ v16bf load_a_frag(const unsigned short* lds,
                                             int row_base, int kk, int lane) {
  int row = row_base + (lane & 15);
  int c0  = kk + ((lane >> 4) << 3);
  const unsigned short* p = lds + row * LDS_STRIDE + c0;
  uint4 lo = *(const uint4*)(p);
  uint4 hi = *(const uint4*)(p + 16);
  Bits256 f;
  f.r[0] = lo.x; f.r[1] = lo.y; f.r[2] = lo.z; f.r[3] = lo.w;
  f.r[4] = hi.x; f.r[5] = hi.y; f.r[6] = hi.z; f.r[7] = hi.w;
  return __builtin_bit_cast(v16bf, f);
}

// B fragment (32x16 bf16): lane n = col; lanes 0-15 hold K[kk+0..7] and K[kk+16..23];
// lanes 16-31 hold K[kk+8..15] and K[kk+24..31].
__device__ __forceinline__ v16bf load_b_frag(const unsigned short* lds,
                                             int col_base, int kk, int lane) {
  int col = col_base + (lane & 15);
  int k0  = kk + ((lane >> 4) << 3);
  const unsigned short* p = lds + col * LDS_STRIDE + k0;
  uint4 lo = *(const uint4*)(p);
  uint4 hi = *(const uint4*)(p + 16);
  Bits256 f;
  f.r[0] = lo.x; f.r[1] = lo.y; f.r[2] = lo.z; f.r[3] = lo.w;
  f.r[4] = hi.x; f.r[5] = hi.y; f.r[6] = hi.z; f.r[7] = hi.w;
  return __builtin_bit_cast(v16bf, f);
}

// ---------------- streaming f32 -> bf16 conversion (optional D_m mask) -----

__global__ __launch_bounds__(256)
void convert_bf16(const float* __restrict__ src,
                  unsigned short* __restrict__ dst,
                  const int* __restrict__ token_mask) {  // null = no mask
  size_t idx = ((size_t)blockIdx.x * 256 + threadIdx.x) * 8;
  int row = (int)(idx >> 11);        // / D_MODEL
  int col = (int)(idx & 2047);
  int dm  = token_mask ? dm_of(token_mask[row]) : D_MODEL;
  const float4 v0 = *(const float4*)(src + idx);
  const float4 v1 = *(const float4*)(src + idx + 4);
  unsigned int p0 = (unsigned int)((col + 0 < dm) ? f2bf(v0.x) : 0) |
                    ((unsigned int)((col + 1 < dm) ? f2bf(v0.y) : 0) << 16);
  unsigned int p1 = (unsigned int)((col + 2 < dm) ? f2bf(v0.z) : 0) |
                    ((unsigned int)((col + 3 < dm) ? f2bf(v0.w) : 0) << 16);
  unsigned int p2 = (unsigned int)((col + 4 < dm) ? f2bf(v1.x) : 0) |
                    ((unsigned int)((col + 5 < dm) ? f2bf(v1.y) : 0) << 16);
  unsigned int p3 = (unsigned int)((col + 6 < dm) ? f2bf(v1.z) : 0) |
                    ((unsigned int)((col + 7 < dm) ? f2bf(v1.w) : 0) << 16);
  uint4 o; o.x = p0; o.y = p1; o.z = p2; o.w = p3;
  *(uint4*)(dst + idx) = o;
}

// ---------------- bf16 WMMA GEMM, pipelined async double buffering ----------
// PASS = 1: epilogue = bias + exact GELU -> bf16 H
// PASS = 2: epilogue = bias + D_m output mask -> f32 Y
template <int PASS>
__global__ __launch_bounds__(NTHREADS)
void nested_ffn_gemm(const unsigned short* __restrict__ A,    // (M,2048) bf16
                     const unsigned short* __restrict__ Wbf,  // (2048,2048) bf16
                     const float* __restrict__ bias,          // (2048,) f32
                     const int*   __restrict__ token_mask,
                     void* __restrict__ Outp) {
  __shared__ unsigned short lds[2][2][BM * LDS_STRIDE];  // [buf][A|B][...]
  __shared__ int sDm[BM];

  const int tid  = threadIdx.x;
  const int lane = tid & 31;
  const int wave = tid >> 5;
  const int wm   = wave >> 1;   // 0..3 -> wave M offset = wm*32
  const int wn   = wave & 1;    // 0..1 -> wave N offset = wn*64

  const int n0 = blockIdx.x * BN;
  const int m0 = blockIdx.y * BM;

  if (PASS == 2 && tid < BM) sDm[tid] = dm_of(token_mask[m0 + tid]);

  const v8f vzero = {0.f, 0.f, 0.f, 0.f, 0.f, 0.f, 0.f, 0.f};
  v8f acc[2][4];
#pragma unroll
  for (int i = 0; i < 2; ++i)
#pragma unroll
    for (int j = 0; j < 4; ++j) acc[i][j] = vzero;

  // Per-thread chunk mapping: tile = 128 rows x 8 chunks of 8 bf16.
  const int c_row  = tid >> 3;         // 0..31 (+32 per it)
  const int c_col8 = (tid & 7) << 3;   // 0,8,...,56

  // Uniform per-block SGPR bases for GVS async loads.
  const unsigned long long baseA =
      (unsigned long long)(__SIZE_TYPE__)(A + (size_t)m0 * D_MODEL);
  const unsigned long long baseB =
      (unsigned long long)(__SIZE_TYPE__)(Wbf + (size_t)n0 * D_MODEL);

  // Loop-invariant per-thread LDS destination addresses (hoisted to VGPRs).
  unsigned int lA_addr[2][4], lB_addr[2][4];
#pragma unroll
  for (int b = 0; b < 2; ++b)
#pragma unroll
    for (int it = 0; it < 4; ++it) {
      unsigned int loff =
          (unsigned int)(((c_row + (it << 5)) * LDS_STRIDE + c_col8) * 2);
      lA_addr[b][it] = (unsigned int)(__SIZE_TYPE__)&lds[b][0][0] + loff;
      lB_addr[b][it] = (unsigned int)(__SIZE_TYPE__)&lds[b][1][0] + loff;
    }

  // Shared 32-bit byte offsets into A / Wbf (advance 2*BK bytes per tile).
  unsigned int voff[4];
#pragma unroll
  for (int it = 0; it < 4; ++it)
    voff[it] = (unsigned int)(((c_row + (it << 5)) * D_MODEL + c_col8) * 2);

  // Issue the 8 per-wave async loads (4 A + 4 B) for the next K-tile.
  auto issue_tile = [&](int b) {
#pragma unroll
    for (int it = 0; it < 4; ++it) {
      async_ld_b128_gvs(lA_addr[b][it], voff[it], baseA);
      async_ld_b128_gvs(lB_addr[b][it], voff[it], baseB);
    }
#pragma unroll
    for (int it = 0; it < 4; ++it) voff[it] += 2u * BK;   // +128 bytes
  };

  auto compute = [&](int b) {
    const unsigned short* lA = &lds[b][0][0];
    const unsigned short* lB = &lds[b][1][0];
#pragma unroll
    for (int kk = 0; kk < BK; kk += 32) {
      v16bf a0 = load_a_frag(lA, wm * 32 + 0,  kk, lane);
      v16bf a1 = load_a_frag(lA, wm * 32 + 16, kk, lane);
      v16bf b0 = load_b_frag(lB, wn * 64 + 0,  kk, lane);
      v16bf b1 = load_b_frag(lB, wn * 64 + 16, kk, lane);
      v16bf b2 = load_b_frag(lB, wn * 64 + 32, kk, lane);
      v16bf b3 = load_b_frag(lB, wn * 64 + 48, kk, lane);
      acc[0][0] = __builtin_amdgcn_wmma_f32_16x16x32_bf16(false, a0, false, b0, (short)0, acc[0][0], false, false);
      acc[0][1] = __builtin_amdgcn_wmma_f32_16x16x32_bf16(false, a0, false, b1, (short)0, acc[0][1], false, false);
      acc[0][2] = __builtin_amdgcn_wmma_f32_16x16x32_bf16(false, a0, false, b2, (short)0, acc[0][2], false, false);
      acc[0][3] = __builtin_amdgcn_wmma_f32_16x16x32_bf16(false, a0, false, b3, (short)0, acc[0][3], false, false);
      acc[1][0] = __builtin_amdgcn_wmma_f32_16x16x32_bf16(false, a1, false, b0, (short)0, acc[1][0], false, false);
      acc[1][1] = __builtin_amdgcn_wmma_f32_16x16x32_bf16(false, a1, false, b1, (short)0, acc[1][1], false, false);
      acc[1][2] = __builtin_amdgcn_wmma_f32_16x16x32_bf16(false, a1, false, b2, (short)0, acc[1][2], false, false);
      acc[1][3] = __builtin_amdgcn_wmma_f32_16x16x32_bf16(false, a1, false, b3, (short)0, acc[1][3], false, false);
    }
  };

  // Software pipeline over NTILES (=32, even) K-tiles, double-buffered.
  issue_tile(0);                              // tile 0 -> buf0
#pragma unroll 1
  for (int t = 0; t < NTILES - 2; t += 2) {
    issue_tile(1);                            // tile t+1 -> buf1
    wait_async_le8();                         // tile t landed (in-order)
    __syncthreads();
    compute(0);
    __syncthreads();
    issue_tile(0);                            // tile t+2 -> buf0
    wait_async_le8();                         // tile t+1 landed
    __syncthreads();
    compute(1);
    __syncthreads();
  }
  issue_tile(1);                              // tile 31 -> buf1
  wait_async_le8();
  __syncthreads();
  compute(0);                                 // tile 30
  __syncthreads();
  wait_async_le0();
  __syncthreads();
  compute(1);                                 // tile 31

  // Epilogue. C/D layout: VGPR v -> row v + 8*(lane>>4); col = lane&15.
#pragma unroll
  for (int i = 0; i < 2; ++i) {
#pragma unroll
    for (int j = 0; j < 4; ++j) {
      int col = n0 + wn * 64 + j * 16 + (lane & 15);
      float bb = bias[col];
#pragma unroll
      for (int v = 0; v < 8; ++v) {
        int lrow = wm * 32 + i * 16 + v + ((lane >> 4) << 3);
        int row  = m0 + lrow;
        float t  = acc[i][j][v] + bb;
        if (PASS == 1) {
          float g = 0.5f * t * (1.0f + erff(t * 0.70710678118654752f));
          ((unsigned short*)Outp)[(size_t)row * D_MODEL + col] = f2bf(g);
        } else {
          float y = (col < sDm[lrow]) ? t : 0.0f;
          ((float*)Outp)[(size_t)row * D_MODEL + col] = y;
        }
      }
    }
  }
}

extern "C" void kernel_launch(void* const* d_in, const int* in_sizes, int n_in,
                              void* d_out, int out_size, void* d_ws, size_t ws_size,
                              hipStream_t stream) {
  // setup_inputs() order: x, w1, b1, w2, b2, token_mask
  const float* x  = (const float*)d_in[0];
  const float* w1 = (const float*)d_in[1];
  const float* b1 = (const float*)d_in[2];
  const float* w2 = (const float*)d_in[3];
  const float* b2 = (const float*)d_in[4];
  const int* token_mask = (const int*)d_in[5];
  float* y = (float*)d_out;

  // Workspace layout (bf16): Xbf 64MiB | W1bf 8MiB | W2bf 8MiB | Hbf 64MiB
  unsigned short* Xbf  = (unsigned short*)d_ws;
  unsigned short* W1bf = Xbf  + (size_t)M_TOTAL * D_MODEL;
  unsigned short* W2bf = W1bf + (size_t)D_MODEL * D_MODEL;
  unsigned short* Hbf  = W2bf + (size_t)D_MODEL * D_MODEL;

  const int nx = (M_TOTAL * D_MODEL) / (8 * 256);   // 16384 blocks
  const int nw = (D_MODEL * D_MODEL) / (8 * 256);   // 2048 blocks
  convert_bf16<<<nx, 256, 0, stream>>>(x,  Xbf,  token_mask);
  convert_bf16<<<nw, 256, 0, stream>>>(w1, W1bf, nullptr);
  convert_bf16<<<nw, 256, 0, stream>>>(w2, W2bf, nullptr);

  dim3 grid(D_MODEL / BN, M_TOTAL / BM);  // (16, 128)
  dim3 block(NTHREADS);
  nested_ffn_gemm<1><<<grid, block, 0, stream>>>(Xbf, W1bf, b1, token_mask,
                                                 (void*)Hbf);
  nested_ffn_gemm<2><<<grid, block, 0, stream>>>(Hbf, W2bf, b2, token_mask,
                                                 (void*)y);
}
